// JujubeCakeCell_50285477102211
// MI455X (gfx1250) — matrix operand order repaired and checked by reading, not compiled, not run on hardware
//
#include <hip/hip_runtime.h>
#include <hip/hip_bf16.h>
#include <math.h>

// ---------------- problem constants ----------------
#define SUB_LSTMS 4
#define SUB_UNITS 256
#define UNITS     1024
#define BATCH     64
#define SEQ       512
#define INPUT_DIM 1024
#define SUB_IN    256

// ---------------- launch shape ----------------
#define NBLK  16
#define NTHR  256

// K-chunk staged in LDS, padded to avoid 16-way bank conflicts
#define KC      256
#define LDSP    (KC + 8)           // 528B row stride: +4 banks/row

typedef __bf16 bf16_t;
typedef bf16_t v16bf __attribute__((ext_vector_type(16)));
typedef float  v8f   __attribute__((ext_vector_type(8)));

union FragAB { v16bf v; uint4 u[2]; };

#define WMMA_BF16(A, B, C) \
    __builtin_amdgcn_wmma_f32_16x16x32_bf16(false, (A), false, (B), (short)0, (C), false, false)

__device__ __forceinline__ float hsig(float x) {
    return fminf(fmaxf(fmaf(0.2f, x, 0.5f), 0.0f), 1.0f);
}

// device-scope phase barrier for a persistent 16-block grid
__device__ __forceinline__ void gbar(unsigned* cnt, unsigned& phase) {
    __builtin_amdgcn_fence(__ATOMIC_RELEASE, "agent");
    __syncthreads();
    ++phase;
    if (threadIdx.x == 0) {
        __hip_atomic_fetch_add(cnt, 1u, __ATOMIC_ACQ_REL, __HIP_MEMORY_SCOPE_AGENT);
        const unsigned target = phase * NBLK;
        while (__hip_atomic_load(cnt, __ATOMIC_ACQUIRE, __HIP_MEMORY_SCOPE_AGENT) < target) {
            __builtin_amdgcn_s_sleep(2);
        }
    }
    __syncthreads();
    __builtin_amdgcn_fence(__ATOMIC_ACQUIRE, "agent");
}

// ---------------- prep kernels ----------------

__global__ void k_convert_x(const float* __restrict__ x, bf16_t* __restrict__ xr) {
    const size_t N = (size_t)SEQ * BATCH * INPUT_DIM;
    for (size_t i = (size_t)blockIdx.x * blockDim.x + threadIdx.x; i < N;
         i += (size_t)gridDim.x * blockDim.x) {
        size_t t = i / (BATCH * INPUT_DIM);
        size_t r = i - t * (BATCH * INPUT_DIM);
        size_t b = r >> 10;
        size_t d = r & 1023;
        xr[i] = (bf16_t)x[(b * SEQ + t) * INPUT_DIM + d];
    }
}

__global__ void k_build_wtc(const float* __restrict__ ck, const float* __restrict__ crk,
                            bf16_t* __restrict__ wtc) {
    const size_t N = (size_t)3 * UNITS * 2048;
    for (size_t i = (size_t)blockIdx.x * blockDim.x + threadIdx.x; i < N;
         i += (size_t)gridDim.x * blockDim.x) {
        size_t n = i >> 11;
        size_t kk = i & 2047;
        float v = (kk < 1024) ? ck[kk * (3 * UNITS) + n]
                              : crk[(kk - 1024) * (3 * UNITS) + n];
        wtc[i] = (bf16_t)v;
    }
}

__global__ void k_build_wts(const float* __restrict__ sk, const float* __restrict__ srk,
                            bf16_t* __restrict__ wts) {
    const size_t N = (size_t)4 * SUB_UNITS * 512;
    for (size_t i = (size_t)blockIdx.x * blockDim.x + threadIdx.x; i < N;
         i += (size_t)gridDim.x * blockDim.x) {
        size_t n = i >> 9;
        size_t kk = i & 511;
        float v = (kk < 256) ? sk[kk * (4 * SUB_UNITS) + n]
                             : srk[(kk - 256) * (4 * SUB_UNITS) + n];
        wts[i] = (bf16_t)v;
    }
}

__global__ void k_init(float* __restrict__ cst, float* __restrict__ scst,
                       float* __restrict__ cnew, bf16_t* __restrict__ hb,
                       bf16_t* __restrict__ shb, unsigned* __restrict__ bar) {
    int i = blockIdx.x * blockDim.x + threadIdx.x;
    if (i < BATCH * UNITS)     { cst[i] = 0.0f; cnew[i] = 0.0f; hb[i] = (bf16_t)0.0f; }
    if (i < BATCH * SUB_UNITS) { scst[i] = 0.0f; shb[i] = (bf16_t)0.0f; }
    if (i == 0) *bar = 0u;
}

// ---------------- persistent scan kernel ----------------

__global__ void __launch_bounds__(NTHR, 1)
scan_kernel(const bf16_t* __restrict__ xr,    // (T,B,D) bf16
            const bf16_t* __restrict__ wtc,   // (3072, 2048) bf16 N-major
            const bf16_t* __restrict__ wts,   // (1024, 512)  bf16 N-major
            const float*  __restrict__ cbias,
            const float*  __restrict__ sbias,
            float* __restrict__ cst,   // 64 x 1024 cake cell state
            float* __restrict__ cnew,  // 64 x 1024 concat of sub cell snapshots
            float* __restrict__ scst,  // 64 x 256  sub cell state
            bf16_t* __restrict__ hb,   // 64 x 1024 bf16 cake h
            bf16_t* __restrict__ shb,  // 64 x 256  bf16 sub h
            float* __restrict__ out,   // (B, T, UNITS) f32
            unsigned* __restrict__ bar)
{
    __shared__ bf16_t lds_a[BATCH][LDSP];   // staged A chunk: 64 rows x 256 (padded)

    const int lane = threadIdx.x & 31;
    const int wid  = blockIdx.x * (NTHR / 32) + (threadIdx.x >> 5);

    const int l15  = lane & 15;
    const int hi   = (lane & 16) ? 1 : 0;
    const int klo  = hi * 16;   // A-frag K sub-offset (contiguous K, matches B)
    const int kb   = hi * 16;   // B-frag K sub-offset (contiguous K, matches A)

    // Visible initialization stores: guarantees lds_a is "written" in compiler
    // view so its loads can never fold to undef (async writes are opaque asm).
    for (int u = threadIdx.x; u < BATCH * LDSP; u += NTHR)
        ((bf16_t*)lds_a)[u] = (bf16_t)0.0f;
    __syncthreads();

    // stage 64 x KC bf16 chunk: CDNA5 async global->LDS copies (ASYNCcnt path).
    // LDS dest offset is the low 32 bits of the generic address of &lds_a[..]
    // (ISA 10.2); taking the address also escapes lds_a.
    auto stage = [&](const bf16_t* src, int gstride) {
#pragma unroll
        for (int j = 0; j < (BATCH * KC / 8) / NTHR; ++j) {   // 8 x 16B per thread
            const int u   = threadIdx.x + j * NTHR;
            const int row = u >> 5;                // 32 16B-units per 256-elem row
            const int c   = (u & 31) * 8;
            const unsigned ldsoff = (unsigned)(uintptr_t)(&lds_a[row][c]);
            const bf16_t* g = src + (size_t)row * gstride + c;
            asm volatile("global_load_async_to_lds_b128 %0, %1, off"
                         :: "v"(ldsoff), "v"(g) : "memory");
        }
    };

    unsigned phase = 0;

    // ---- sub GEMM: 4 waves/block active; slot = (mt, n-group); 4 gate chains ----
    const int w_in_b = threadIdx.x >> 5;           // 0..7
    const bool s_act = (w_in_b < 4);
    const int s_slot = blockIdx.x * 4 + w_in_b;    // 0..63
    const int s_mt   = (s_slot >> 4) & 3;          // 0..3
    const int s_g    = s_slot & 15;                // 0..15
    const int s_row  = s_mt * 16 + l15;
    const int s_n    = s_g * 16 + l15;             // 0..255
    const bf16_t* s_bwI = wts + (size_t)(0 * SUB_UNITS + s_n) * 512;
    const bf16_t* s_bwF = wts + (size_t)(1 * SUB_UNITS + s_n) * 512;
    const bf16_t* s_bwG = wts + (size_t)(2 * SUB_UNITS + s_n) * 512;
    const bf16_t* s_bwO = wts + (size_t)(3 * SUB_UNITS + s_n) * 512;

    // ---- cake GEMM: all 128 waves; wave = (m-half, n-group); 3 gates x 2 mt ----
    const int c_g  = wid & 63;                     // 0..63
    const int c_mh = wid >> 6;                     // 0..1
    const int c_r0 = c_mh * 32 + l15;
    const int c_r1 = c_r0 + 16;
    const int c_n  = c_g * 16 + l15;               // 0..1023
    const bf16_t* c_bwI = wtc + (size_t)(c_n)        * 2048;
    const bf16_t* c_bwF = wtc + (size_t)(1024 + c_n) * 2048;
    const bf16_t* c_bwO = wtc + (size_t)(2048 + c_n) * 2048;

    for (int t = 0; t < SEQ; ++t) {
        const bf16_t* xt = xr + (size_t)t * (BATCH * INPUT_DIM);

        if (blockIdx.x == 0 && threadIdx.x < BATCH && (t + 1) < SEQ)
            __builtin_prefetch(xt + BATCH * INPUT_DIM + threadIdx.x * INPUT_DIM, 0, 1);

        // ======== four sequential sub-LSTM steps ========
        for (int ks = 0; ks < SUB_LSTMS; ++ks) {
            v8f aI = {}, aF = {}, aG = {}, aO = {};
#pragma unroll
            for (int kc = 0; kc < 512; kc += KC) {    // chunk0: x-slice, chunk1: sh
                __syncthreads();                      // LDS reads of prev chunk done
                if (kc == 0) stage(xt + ks * SUB_IN, INPUT_DIM);
                else         stage(shb, SUB_UNITS);
                asm volatile("s_wait_asynccnt 0x0" ::: "memory");
                __syncthreads();
                if (s_act) {
#pragma unroll
                    for (int kk = 0; kk < KC; kk += 32) {
                        FragAB a, bI, bF, bG, bO;
                        a.u[0] = *(const uint4*)(&lds_a[s_row][kk + klo]);
                        a.u[1] = *(const uint4*)(&lds_a[s_row][kk + 8 + klo]);
                        const bf16_t* pI = s_bwI + kc + kk + kb;
                        const bf16_t* pF = s_bwF + kc + kk + kb;
                        const bf16_t* pG = s_bwG + kc + kk + kb;
                        const bf16_t* pO = s_bwO + kc + kk + kb;
                        bI.u[0] = *(const uint4*)(pI); bI.u[1] = *(const uint4*)(pI + 8);
                        bF.u[0] = *(const uint4*)(pF); bF.u[1] = *(const uint4*)(pF + 8);
                        bG.u[0] = *(const uint4*)(pG); bG.u[1] = *(const uint4*)(pG + 8);
                        bO.u[0] = *(const uint4*)(pO); bO.u[1] = *(const uint4*)(pO + 8);
                        aI = WMMA_BF16(a.v, bI.v, aI);   // 4 independent chains
                        aF = WMMA_BF16(a.v, bF.v, aF);
                        aG = WMMA_BF16(a.v, bG.v, aG);
                        aO = WMMA_BF16(a.v, bO.v, aO);
                    }
                }
            }
            if (s_act) {
                // fused gates: nonlinearity directly on accumulators
                const float bI = sbias[s_n];
                const float bF = sbias[SUB_UNITS + s_n];
                const float bG = sbias[2 * SUB_UNITS + s_n];
                const float bO = sbias[3 * SUB_UNITS + s_n];
                const int mb = s_mt * 16 + hi * 8;
#pragma unroll
                for (int i2 = 0; i2 < 8; ++i2) {
                    const int b = mb + i2;
                    const size_t si = (size_t)b * SUB_UNITS + s_n;
                    float ig = hsig(aI[i2] + bI);
                    float fg = hsig(aF[i2] + bF);
                    float og = hsig(aO[i2] + bO);
                    float cc = fg * scst[si] + ig * tanhf(aG[i2] + bG);
                    float hh = og * tanhf(cc);
                    scst[si] = cc;
                    cnew[(size_t)b * UNITS + ks * SUB_UNITS + s_n] = cc;
                    shb[si] = (bf16_t)hh;
                }
            }
            gbar(bar, phase);
        }

        // ======== cake: A=[x_t | h] (64x2048) @ wtc^T, gates fused ========
        {
            v8f aI0 = {}, aI1 = {}, aF0 = {}, aF1 = {}, aO0 = {}, aO1 = {};
#pragma unroll
            for (int kc = 0; kc < 2048; kc += KC) {   // 4 chunks x_t, 4 chunks h
                __syncthreads();
                if (kc < 1024) stage(xt + kc, INPUT_DIM);
                else           stage(hb + (kc - 1024), UNITS);
                asm volatile("s_wait_asynccnt 0x0" ::: "memory");
                __syncthreads();
#pragma unroll 4
                for (int kk = 0; kk < KC; kk += 32) {
                    FragAB fa0, fa1, bI, bF, bO;
                    fa0.u[0] = *(const uint4*)(&lds_a[c_r0][kk + klo]);
                    fa0.u[1] = *(const uint4*)(&lds_a[c_r0][kk + 8 + klo]);
                    fa1.u[0] = *(const uint4*)(&lds_a[c_r1][kk + klo]);
                    fa1.u[1] = *(const uint4*)(&lds_a[c_r1][kk + 8 + klo]);
                    const bf16_t* pI = c_bwI + kc + kk + kb;
                    const bf16_t* pF = c_bwF + kc + kk + kb;
                    const bf16_t* pO = c_bwO + kc + kk + kb;
                    bI.u[0] = *(const uint4*)(pI); bI.u[1] = *(const uint4*)(pI + 8);
                    bF.u[0] = *(const uint4*)(pF); bF.u[1] = *(const uint4*)(pF + 8);
                    bO.u[0] = *(const uint4*)(pO); bO.u[1] = *(const uint4*)(pO + 8);
                    aI0 = WMMA_BF16(fa0.v, bI.v, aI0);   // 6 independent chains
                    aI1 = WMMA_BF16(fa1.v, bI.v, aI1);
                    aF0 = WMMA_BF16(fa0.v, bF.v, aF0);
                    aF1 = WMMA_BF16(fa1.v, bF.v, aF1);
                    aO0 = WMMA_BF16(fa0.v, bO.v, aO0);
                    aO1 = WMMA_BF16(fa1.v, bO.v, aO1);
                }
            }
            const float bI = cbias[c_n];
            const float bF = cbias[UNITS + c_n];
            const float bO = cbias[2 * UNITS + c_n];
            const int mb0 = c_mh * 32 + hi * 8;
#pragma unroll
            for (int i2 = 0; i2 < 8; ++i2) {
                {
                    const int b = mb0 + i2;
                    const size_t ci = (size_t)b * UNITS + c_n;
                    float ig = hsig(aI0[i2] + bI);
                    float fg = hsig(aF0[i2] + bF);
                    float og = hsig(aO0[i2] + bO);
                    float c2 = fg * cst[ci] + ig * tanhf(cnew[ci]);
                    float h2 = og * tanhf(c2);
                    cst[ci] = c2;
                    hb[ci] = (bf16_t)h2;
                    out[(size_t)b * SEQ * UNITS + (size_t)t * UNITS + c_n] = h2;
                }
                {
                    const int b = mb0 + 16 + i2;
                    const size_t ci = (size_t)b * UNITS + c_n;
                    float ig = hsig(aI1[i2] + bI);
                    float fg = hsig(aF1[i2] + bF);
                    float og = hsig(aO1[i2] + bO);
                    float c2 = fg * cst[ci] + ig * tanhf(cnew[ci]);
                    float h2 = og * tanhf(c2);
                    cst[ci] = c2;
                    hb[ci] = (bf16_t)h2;
                    out[(size_t)b * SEQ * UNITS + (size_t)t * UNITS + c_n] = h2;
                }
            }
        }
        gbar(bar, phase);
    }
}

// ---------------- host side ----------------

extern "C" void kernel_launch(void* const* d_in, const int* in_sizes, int n_in,
                              void* d_out, int out_size, void* d_ws, size_t ws_size,
                              hipStream_t stream) {
    (void)in_sizes; (void)n_in; (void)out_size; (void)ws_size;
    const float* x   = (const float*)d_in[0];
    const float* ck  = (const float*)d_in[1];
    const float* crk = (const float*)d_in[2];
    const float* cb  = (const float*)d_in[3];
    const float* sk  = (const float*)d_in[4];
    const float* srk = (const float*)d_in[5];
    const float* sb  = (const float*)d_in[6];
    float* out = (float*)d_out;

    char* ws = (char*)d_ws;
    auto alloc = [&](size_t bytes) -> char* {
        char* p = ws;
        ws += (bytes + 255) & ~(size_t)255;
        return p;
    };
    bf16_t* xr  = (bf16_t*)alloc((size_t)SEQ * BATCH * INPUT_DIM * sizeof(bf16_t));
    bf16_t* wtc = (bf16_t*)alloc((size_t)3 * UNITS * 2048 * sizeof(bf16_t));
    bf16_t* wts = (bf16_t*)alloc((size_t)4 * SUB_UNITS * 512 * sizeof(bf16_t));
    float*  cst = (float*) alloc((size_t)BATCH * UNITS * sizeof(float));
    float*  cnw = (float*) alloc((size_t)BATCH * UNITS * sizeof(float));
    float*  scs = (float*) alloc((size_t)BATCH * SUB_UNITS * sizeof(float));
    bf16_t* hb  = (bf16_t*)alloc((size_t)BATCH * UNITS * sizeof(bf16_t));
    bf16_t* shb = (bf16_t*)alloc((size_t)BATCH * SUB_UNITS * sizeof(bf16_t));
    unsigned* bar = (unsigned*)alloc(256);

    k_convert_x<<<dim3(2048), dim3(256), 0, stream>>>(x, xr);
    k_build_wtc<<<dim3(1024), dim3(256), 0, stream>>>(ck, crk, wtc);
    k_build_wts<<<dim3(256),  dim3(256), 0, stream>>>(sk, srk, wts);
    k_init<<<dim3((BATCH * UNITS + 255) / 256), dim3(256), 0, stream>>>(
        cst, scs, cnw, hb, shb, bar);

    scan_kernel<<<dim3(NBLK), dim3(NTHR), 0, stream>>>(
        xr, wtc, wts, cb, sb, cst, cnw, scs, hb, shb, out, bar);
}
